// VisionAttention_5497558139049
// MI455X (gfx1250) — matrix operand order repaired and checked by reading, not compiled, not run on hardware
//
#include <hip/hip_runtime.h>
#include <hip/hip_bf16.h>

#define HIDDEN 1280
#define SEQ    4096
#define NH     16
#define HD     80
#define HDP    96   // head dim padded to 3x32 for K=32 WMMA steps

typedef __attribute__((ext_vector_type(4)))  _Float16 v4h;
typedef __attribute__((ext_vector_type(8)))  _Float16 v8h;
typedef __attribute__((ext_vector_type(16))) _Float16 v16h;
typedef __attribute__((ext_vector_type(8)))  float    v8f;
typedef __attribute__((ext_vector_type(4)))  unsigned int u32x4;
typedef __attribute__((ext_vector_type(8)))  int i32x8;
typedef __attribute__((ext_vector_type(4)))  int i32x4;

static constexpr float SCALING_F = 0.1118033988749895f; // 80^-0.5

// ---------------------------------------------------------------- fragments
__device__ __forceinline__ v16h cat16(v8h lo, v8h hi) {
  return __builtin_shufflevector(lo, hi, 0,1,2,3,4,5,6,7,8,9,10,11,12,13,14,15);
}
// A-matrix 16x32 f16 fragment: lane holds row = lane&15,
// halves K = i + (lane>>4)*16  -> two contiguous 16B chunks.
__device__ __forceinline__ v16h load_afrag(const _Float16* rowp, int kh) {
  v8h lo = *(const v8h*)(rowp + kh);
  v8h hi = *(const v8h*)(rowp + kh + 8);
  return cat16(lo, hi);
}
// B-matrix 32x16 f16 fragment: lane holds col = lane&15,
// K = (lane>>4)*16 .. +15 contiguous (pointer pre-offset by that base).
__device__ __forceinline__ v16h load_bfrag(const _Float16* colp) {
  v8h lo = *(const v8h*)(colp);
  v8h hi = *(const v8h*)(colp + 8);
  return cat16(lo, hi);
}
__device__ __forceinline__ v8f wmma16(v16h a, v16h b, v8f c) {
  return __builtin_amdgcn_wmma_f32_16x16x32_f16(false, a, false, b, (short)0, c,
                                                false, false);
}

// ---------------------------------------------------------------- TDM
// Issue one Tensor-Data-Mover 2-D tile load: tile_d1 rows x tile_d0 halves,
// row stride `stride0` (halves) in global, padded in LDS by 8 halves per
// 32-half row (pad_interval=16 DW, pad_amount=4 DW) -> LDS row stride 40.
// D# layout per CDNA5 ISA ch.8 (group0: 128b, group1: 256b; groups2/3 zero).
__device__ __forceinline__ void tdm_load_tile(const _Float16* gsrc,
                                              unsigned int lds_off,
                                              int tensor_d0, int tensor_d1,
                                              int tile_d0, int tile_d1,
                                              int stride0) {
  unsigned long long ga = (unsigned long long)(uintptr_t)gsrc;
  u32x4 g0;
  g0[0] = 1u;                                   // count=1, user descriptor
  g0[1] = lds_off;                              // lds_addr (bytes)
  g0[2] = (unsigned int)(ga & 0xFFFFFFFFu);     // global_addr[31:0]
  g0[3] = (unsigned int)((ga >> 32) & 0x1FFFFFFu) | (2u << 30); // [56:32] | type=2
  i32x8 g1;
  g1[0] = (1 << 16)    // data_size = 2 bytes
        | (1 << 20)    // pad_enable
        | (3 << 22)    // pad_interval code 3 = 16 DWORDs (one 32-half row)
        | (3 << 25);   // pad_amount  code 3 = 4 DWORDs (8 halves)
  g1[1] = (tensor_d0 & 0xFFFF) << 16;                              // dim0 lo16
  g1[2] = ((tensor_d0 >> 16) & 0xFFFF) | ((tensor_d1 & 0xFFFF) << 16);
  g1[3] = ((tensor_d1 >> 16) & 0xFFFF) | ((tile_d0 & 0xFFFF) << 16);
  g1[4] = (tile_d1 & 0xFFFF);                                      // tile_dim2=0
  g1[5] = stride0;                                                 // dim0 stride
  g1[6] = 0;
  g1[7] = 0;
  i32x4 z4 = {0, 0, 0, 0};
#if __clang_major__ >= 23
  i32x8 z8 = {0, 0, 0, 0, 0, 0, 0, 0};
  __builtin_amdgcn_tensor_load_to_lds(g0, g1, z4, z4, z8, 0);
#else
  __builtin_amdgcn_tensor_load_to_lds(g0, g1, z4, z4, 0);
#endif
}

// ---------------------------------------------------------------- converts
__global__ __launch_bounds__(256)
void f32_to_f16_kernel(const float* __restrict__ in, _Float16* __restrict__ out, int n) {
  int i = (blockIdx.x * blockDim.x + threadIdx.x) * 4;
  if (i + 3 < n) {
    float4 f = *(const float4*)(in + i);
    v4h h;
    h.x = (_Float16)f.x; h.y = (_Float16)f.y;
    h.z = (_Float16)f.z; h.w = (_Float16)f.w;
    *(v4h*)(out + i) = h;
  }
}

// W[K][N] f32  ->  WT[N][K] f16  (32x32 LDS tile transpose)
__global__ __launch_bounds__(256)
void transpose_w_kernel(const float* __restrict__ W, _Float16* __restrict__ WT,
                        int K, int N) {
  __shared__ float tile[32][33];
  const int n0 = blockIdx.x * 32;
  const int k0 = blockIdx.y * 32;
  const int tx = threadIdx.x, ty = threadIdx.y; // (32,8)
#pragma unroll
  for (int i = 0; i < 4; ++i)
    tile[ty + i * 8][tx] = W[(size_t)(k0 + ty + i * 8) * N + n0 + tx];
  __syncthreads();
#pragma unroll
  for (int i = 0; i < 4; ++i)
    WT[(size_t)(n0 + ty + i * 8) * K + k0 + tx] = (_Float16)tile[tx][ty + i * 8];
}

// ---------------------------------------------------------------- WMMA GEMM
// C[M][N] = A[M][K] (f16, row-major) @ BT[N][K] (f16) + bias[N] (f32)
// block = 256 thr (8 waves), tile 128x128, k-step 32; wave -> 32x64 subtile.
// Tiles are DMA'd into LDS by the TDM (double buffered); wave 0 issues the
// descriptors and gates the block with s_wait_tensorcnt + barrier.
template <bool OUT_F32>
__global__ __launch_bounds__(256)
void gemm_wmma_kernel(const _Float16* __restrict__ A, const _Float16* __restrict__ BT,
                      const float* __restrict__ bias, void* __restrict__ Cout,
                      int M, int N, int K) {
  constexpr int LDT = 40; // 32 + 8 halves TDM pad -> conflict-free b128 reads
  __shared__ __align__(16) _Float16 As[2][128 * LDT];
  __shared__ __align__(16) _Float16 Bs[2][128 * LDT];

  const int m0 = blockIdx.y * 128;
  const int n0 = blockIdx.x * 128;
  const int t = threadIdx.x;
  const int wave = t >> 5, lane = t & 31;
  const int lh = lane & 15, hb = lane >> 4;
  const int wm = wave & 3;  // 32-row strip
  const int wn = wave >> 2; // 64-col strip

  const _Float16* Abase = A + (size_t)m0 * K;
  const _Float16* Bbase = BT + (size_t)n0 * K;
  const unsigned int asOff[2] = {(unsigned int)(uintptr_t)&As[0][0],
                                 (unsigned int)(uintptr_t)&As[1][0]};
  const unsigned int bsOff[2] = {(unsigned int)(uintptr_t)&Bs[0][0],
                                 (unsigned int)(uintptr_t)&Bs[1][0]};

  v8f acc[2][4];
#pragma unroll
  for (int mi = 0; mi < 2; ++mi)
#pragma unroll
    for (int ni = 0; ni < 4; ++ni) acc[mi][ni] = {};

  if (wave == 0) { // prologue: DMA first k-tile pair into buffer 0
    tdm_load_tile(Abase, asOff[0], K, M, 32, 128, K);
    tdm_load_tile(Bbase, bsOff[0], K, N, 32, 128, K);
  }

  for (int k0 = 0, it = 0; k0 < K; k0 += 32, ++it) {
    const int cur = it & 1;
    if (wave == 0) {
      if (k0 + 32 < K) { // kick off next tile pair into the other buffer
        tdm_load_tile(Abase + k0 + 32, asOff[cur ^ 1], K, M, 32, 128, K);
        tdm_load_tile(Bbase + k0 + 32, bsOff[cur ^ 1], K, N, 32, 128, K);
        __builtin_amdgcn_s_wait_tensorcnt(2); // current pair done, next in flight
      } else {
        __builtin_amdgcn_s_wait_tensorcnt(0);
      }
    }
    __syncthreads(); // tile `cur` visible to all waves

    v16h af[2], bf[4];
#pragma unroll
    for (int mi = 0; mi < 2; ++mi)
      af[mi] = load_afrag(&As[cur][(wm * 32 + mi * 16 + lh) * LDT], hb * 16);
#pragma unroll
    for (int ni = 0; ni < 4; ++ni)
      bf[ni] = load_bfrag(&Bs[cur][(wn * 64 + ni * 16 + lh) * LDT + hb * 16]);
#pragma unroll
    for (int mi = 0; mi < 2; ++mi)
#pragma unroll
      for (int ni = 0; ni < 4; ++ni)
        acc[mi][ni] = wmma16(af[mi], bf[ni], acc[mi][ni]);

    __syncthreads(); // all waves done reading `cur` before TDM overwrites it
  }

#pragma unroll
  for (int mi = 0; mi < 2; ++mi) {
#pragma unroll
    for (int ni = 0; ni < 4; ++ni) {
      const int col = n0 + wn * 64 + ni * 16 + lh;
      const float bv = bias[col];
#pragma unroll
      for (int vi = 0; vi < 8; ++vi) {
        const int row = m0 + wm * 32 + mi * 16 + hb * 8 + vi; // C-layout row
        const float val = acc[mi][ni][vi] + bv;
        if (OUT_F32)
          ((float*)Cout)[(size_t)row * N + col] = val;
        else
          ((_Float16*)Cout)[(size_t)row * N + col] = (_Float16)val;
      }
    }
  }
}

// ---------------------------------------------------------------- RoPE+pack
// qkv f16 [SEQ][3*HIDDEN] -> Qp,Kp [NH][SEQ][HDP] (RoPE'd, zero-padded),
//                            Vt    [NH][HDP][SEQ] (transposed for B-frags).
__global__ __launch_bounds__(256)
void rope_pack_kernel(const _Float16* __restrict__ qkv, const float* __restrict__ cosb,
                      const float* __restrict__ sinb, _Float16* __restrict__ Qp,
                      _Float16* __restrict__ Kp, _Float16* __restrict__ Vt) {
  const int idx = blockIdx.x * 256 + threadIdx.x; // (s,h)
  const int s = idx >> 4, h = idx & 15;
  const _Float16* q = qkv + (size_t)s * 3 * HIDDEN + h * HD;
  const _Float16* k = q + HIDDEN;
  const _Float16* v = k + HIDDEN;
  const float* cs = cosb + (size_t)s * HD;
  const float* sn = sinb + (size_t)s * HD;
  _Float16* qo = Qp + ((size_t)h * SEQ + s) * HDP;
  _Float16* ko = Kp + ((size_t)h * SEQ + s) * HDP;
#pragma unroll 4
  for (int d = 0; d < HD; ++d) {
    const float c = cs[d], si = sn[d];
    const float qv = (float)q[d], kv = (float)k[d];
    const float qr = (d < HD / 2) ? -(float)q[d + HD / 2] : (float)q[d - HD / 2];
    const float kr = (d < HD / 2) ? -(float)k[d + HD / 2] : (float)k[d - HD / 2];
    qo[d] = (_Float16)(qv * c + qr * si);
    ko[d] = (_Float16)(kv * c + kr * si);
    Vt[((size_t)h * HDP + d) * SEQ + s] = v[d];
  }
  for (int d = HD; d < HDP; ++d) {
    qo[d] = (_Float16)0.f;
    ko[d] = (_Float16)0.f;
    Vt[((size_t)h * HDP + d) * SEQ + s] = (_Float16)0.f;
  }
}

// ---------------------------------------------------------------- flash attn
// One wave handles 16 query rows of one head; streams all 4096 keys in 32-key
// steps with online softmax. Scores never touch memory.
__global__ __launch_bounds__(32)
void flash_attn_kernel(const _Float16* __restrict__ Qp, const _Float16* __restrict__ Kp,
                       const _Float16* __restrict__ Vt, const float* __restrict__ mask,
                       _Float16* __restrict__ Oout) {
  constexpr int LDP = 40;
  __shared__ __align__(16) _Float16 Ptile[16 * LDP]; // C-layout -> A-layout bridge
  const int h = blockIdx.y;
  const int q0 = blockIdx.x * 16;
  const int lane = threadIdx.x;
  const int lh = lane & 15, hb = lane >> 4;

  const _Float16* Qh = Qp + (size_t)h * SEQ * HDP;
  const _Float16* Kh = Kp + (size_t)h * SEQ * HDP;
  const _Float16* Vh = Vt + (size_t)h * HDP * SEQ;

  v16h qf[3];
#pragma unroll
  for (int c = 0; c < 3; ++c)
    qf[c] = load_afrag(Qh + (size_t)(q0 + lh) * HDP + c * 32, hb * 16);

  v8f o[5];
#pragma unroll
  for (int n = 0; n < 5; ++n) o[n] = {};
  float m_i[8], l_i[8];
#pragma unroll
  for (int vi = 0; vi < 8; ++vi) { m_i[vi] = -1e30f; l_i[vi] = 0.f; }

  for (int kb = 0; kb < SEQ; kb += 32) {
    if (kb + 32 < SEQ) { // stream hints for the next key/value block
      __builtin_prefetch(Kh + (size_t)(kb + 32 + lh) * HDP, 0, 0);
      __builtin_prefetch(Vh + (size_t)lh * SEQ + kb + 32, 0, 0);
    }
    v8f s0 = {}, s1 = {};
#pragma unroll
    for (int c = 0; c < 3; ++c) {
      v16h kf = load_bfrag(Kh + (size_t)(kb + lh) * HDP + c * 32 + hb * 16);
      s0 = wmma16(qf[c], kf, s0);
    }
#pragma unroll
    for (int c = 0; c < 3; ++c) {
      v16h kf = load_bfrag(Kh + (size_t)(kb + 16 + lh) * HDP + c * 32 + hb * 16);
      s1 = wmma16(qf[c], kf, s1);
    }

    float p0[8], p1[8];
#pragma unroll
    for (int vi = 0; vi < 8; ++vi) {
      const int row = q0 + hb * 8 + vi;
      float x0 = s0[vi] * SCALING_F + mask[(size_t)row * SEQ + kb + lh];
      float x1 = s1[vi] * SCALING_F + mask[(size_t)row * SEQ + kb + 16 + lh];
      float v = fmaxf(x0, x1); // row reduce within 16-lane C-layout group
      v = fmaxf(v, __shfl_xor(v, 1, 32));
      v = fmaxf(v, __shfl_xor(v, 2, 32));
      v = fmaxf(v, __shfl_xor(v, 4, 32));
      v = fmaxf(v, __shfl_xor(v, 8, 32));
      const float mn = fmaxf(m_i[vi], v);
      const float alpha = __expf(m_i[vi] - mn);
      const float e0 = __expf(x0 - mn), e1 = __expf(x1 - mn);
      float rs = e0 + e1;
      rs += __shfl_xor(rs, 1, 32);
      rs += __shfl_xor(rs, 2, 32);
      rs += __shfl_xor(rs, 4, 32);
      rs += __shfl_xor(rs, 8, 32);
      l_i[vi] = l_i[vi] * alpha + rs;
      m_i[vi] = mn;
      p0[vi] = e0; p1[vi] = e1;
#pragma unroll
      for (int n = 0; n < 5; ++n) o[n][vi] *= alpha;
    }

#pragma unroll
    for (int vi = 0; vi < 8; ++vi) {
      Ptile[(hb * 8 + vi) * LDP + lh] = (_Float16)p0[vi];
      Ptile[(hb * 8 + vi) * LDP + 16 + lh] = (_Float16)p1[vi];
    }
    __syncthreads();
    v16h pf = load_afrag(&Ptile[lh * LDP], hb * 16);
#pragma unroll
    for (int n = 0; n < 5; ++n) {
      v16h vf = load_bfrag(Vh + (size_t)(n * 16 + lh) * SEQ + kb + hb * 16);
      o[n] = wmma16(pf, vf, o[n]);
    }
    __syncthreads();
  }

#pragma unroll
  for (int n = 0; n < 5; ++n)
#pragma unroll
    for (int vi = 0; vi < 8; ++vi) {
      const int row = q0 + hb * 8 + vi;
      Oout[(size_t)row * HIDDEN + h * HD + n * 16 + lh] =
          (_Float16)(o[n][vi] / l_i[vi]);
    }
}

// ---------------------------------------------------------------- launcher
extern "C" void kernel_launch(void* const* d_in, const int* in_sizes, int n_in,
                              void* d_out, int out_size, void* d_ws, size_t ws_size,
                              hipStream_t stream) {
  const float* hid   = (const float*)d_in[0];
  const float* mask  = (const float*)d_in[1];
  const float* cosb  = (const float*)d_in[2];
  const float* sinb  = (const float*)d_in[3];
  const float* Wqkv  = (const float*)d_in[4];
  const float* bqkv  = (const float*)d_in[5];
  const float* Wproj = (const float*)d_in[6];
  const float* bproj = (const float*)d_in[7];

  char* ws = (char*)d_ws;
  size_t off = 0;
  auto take = [&](size_t bytes) -> char* {
    char* p = ws + off;
    off += (bytes + 255) & ~size_t(255);
    return p;
  };
  _Float16* Ah     = (_Float16*)take((size_t)SEQ * HIDDEN * 2); // hidden f16, later reused as attn-out
  _Float16* WqkvT  = (_Float16*)take((size_t)3 * HIDDEN * HIDDEN * 2);
  _Float16* WprojT = (_Float16*)take((size_t)HIDDEN * HIDDEN * 2);
  _Float16* qkvh   = (_Float16*)take((size_t)SEQ * 3 * HIDDEN * 2);
  _Float16* Qp     = (_Float16*)take((size_t)NH * SEQ * HDP * 2);
  _Float16* Kp     = (_Float16*)take((size_t)NH * SEQ * HDP * 2);
  _Float16* Vt     = (_Float16*)take((size_t)NH * HDP * SEQ * 2);

  // 1) hidden f32 -> f16
  f32_to_f16_kernel<<<dim3((SEQ * HIDDEN / 4 + 255) / 256), 256, 0, stream>>>(
      hid, Ah, SEQ * HIDDEN);
  // 2) weight transpose+convert: W[K][N] f32 -> WT[N][K] f16
  transpose_w_kernel<<<dim3(3 * HIDDEN / 32, HIDDEN / 32), dim3(32, 8), 0, stream>>>(
      Wqkv, WqkvT, HIDDEN, 3 * HIDDEN);
  transpose_w_kernel<<<dim3(HIDDEN / 32, HIDDEN / 32), dim3(32, 8), 0, stream>>>(
      Wproj, WprojT, HIDDEN, HIDDEN);
  // 3) qkv = hidden @ Wqkv + bqkv  (f16 out, TDM-staged tiles)
  gemm_wmma_kernel<false><<<dim3(3 * HIDDEN / 128, SEQ / 128), 256, 0, stream>>>(
      Ah, WqkvT, bqkv, qkvh, SEQ, 3 * HIDDEN, HIDDEN);
  // 4) RoPE + head packing (Q/K padded 80->96, V transposed)
  rope_pack_kernel<<<dim3(SEQ * NH / 256), 256, 0, stream>>>(qkvh, cosb, sinb, Qp,
                                                             Kp, Vt);
  // 5) flash attention -> attn-out f16 (aliases Ah, whose data is now dead)
  flash_attn_kernel<<<dim3(SEQ / 16, NH), 32, 0, stream>>>(Qp, Kp, Vt, mask, Ah);
  // 6) out = attn @ Wproj + bproj (f32 out, TDM-staged tiles)
  gemm_wmma_kernel<true><<<dim3(HIDDEN / 128, SEQ / 128), 256, 0, stream>>>(
      Ah, WprojT, bproj, d_out, SEQ, HIDDEN, HIDDEN);
}
